// EncoderSelfAttention_29102698398183
// MI455X (gfx1250) — matrix operand order repaired and checked by reading, not compiled, not run on hardware
//
#include <hip/hip_runtime.h>
#include <hip/hip_bf16.h>
#include <cstdint>

#define BB   32
#define NQS  1024
#define CCH  256
#define MMR  (BB * NQS)       // 32768 rows
#define PPD  8
#define HID  64
#define ZDIM 32
#define EPSF 1e-5f
#define ATT_SCALE 0.0625f     // 1/sqrt(256)

typedef __attribute__((ext_vector_type(16))) _Float16 v16h;
typedef __attribute__((ext_vector_type(8)))  float    v8f;

union Frag { v16h v; uint4 u[2]; _Float16 h[16]; };
union H8   { uint4 u; _Float16 h[8]; };

static __device__ __forceinline__ v8f wmma_f16(const Frag& a, const Frag& b, v8f c) {
  return __builtin_amdgcn_wmma_f32_16x16x32_f16(false, a.v, false, b.v, (short)0, c,
                                                false, false);
}

// ---------------------------------------------------------------------------
// Generic GEMM: Y = act(X(MxK) * W(NxK)^T + bias), optional row-mask zeroing.
// 128x64 tile / workgroup, 8 waves, each wave owns 16 rows x 64 cols.
// ---------------------------------------------------------------------------
__global__ __launch_bounds__(256)
void gemm_wmma_kernel(const float* __restrict__ X, const float* __restrict__ W,
                      const float* __restrict__ bias,
                      const unsigned char* __restrict__ mask,
                      float* __restrict__ Yf, _Float16* __restrict__ Yh,
                      int M, int N, int K, int relu)
{
  __shared__ alignas(16) _Float16 Xs[128][40];  // 32-wide K chunk, pad to 40
  __shared__ alignas(16) _Float16 Ws[64][40];

  const int tid   = threadIdx.x;
  const int wv    = tid >> 5;
  const int lane  = tid & 31;
  const int lhalf = lane & 15;
  const int lhi   = lane >> 4;
  const int nBase = blockIdx.x * 64;
  const int mBase = blockIdx.y * 128;

  v8f z = {};
  v8f acc[4];
#pragma unroll
  for (int t = 0; t < 4; ++t) acc[t] = z;

  for (int kk = 0; kk < K; kk += 32) {
    // stage X tile (128x32 f32 -> f16)
#pragma unroll
    for (int it = 0; it < 4; ++it) {
      int e4 = tid + it * 256;                 // float4 index within tile
      int r = e4 >> 3, c = (e4 & 7) * 4;
      float4 f = *reinterpret_cast<const float4*>(&X[(size_t)(mBase + r) * K + kk + c]);
      Xs[r][c + 0] = (_Float16)f.x; Xs[r][c + 1] = (_Float16)f.y;
      Xs[r][c + 2] = (_Float16)f.z; Xs[r][c + 3] = (_Float16)f.w;
    }
    // stage W tile (64x32 f32 -> f16)
#pragma unroll
    for (int it = 0; it < 2; ++it) {
      int e4 = tid + it * 256;
      int r = e4 >> 3, c = (e4 & 7) * 4;
      float4 f = *reinterpret_cast<const float4*>(&W[(size_t)(nBase + r) * K + kk + c]);
      Ws[r][c + 0] = (_Float16)f.x; Ws[r][c + 1] = (_Float16)f.y;
      Ws[r][c + 2] = (_Float16)f.z; Ws[r][c + 3] = (_Float16)f.w;
    }
    __syncthreads();

    // A fragment: 16 rows x 32 K (ISA 16-bit A layout: two 16B chunks / lane)
    Frag a;
    {
      const uint4* p = reinterpret_cast<const uint4*>(&Xs[wv * 16 + lhalf][lhi * 8]);
      a.u[0] = p[0]; a.u[1] = p[2];
    }
#pragma unroll
    for (int nt = 0; nt < 4; ++nt) {
      Frag b;   // B layout: lane = column, K = 8*(lane>=16)+{0..7, 16..23} (same as A)
      const uint4* p = reinterpret_cast<const uint4*>(&Ws[nt * 16 + lhalf][lhi * 8]);
      b.u[0] = p[0]; b.u[1] = p[2];
      acc[nt] = wmma_f16(a, b, acc[nt]);
    }
    __syncthreads();
  }

  // ---- epilogue: branchless mask/relu, hoisted mask bytes, hoisted guards
  const int rowBase = mBase + wv * 16 + lhi * 8;   // D layout rows
  float keep[8];
#pragma unroll
  for (int v = 0; v < 8; ++v) keep[v] = 1.f;
  if (mask) {
#pragma unroll
    for (int v = 0; v < 8; ++v) keep[v] = mask[rowBase + v] ? 1.f : 0.f;
  }
#pragma unroll
  for (int nt = 0; nt < 4; ++nt) {
    int n = nBase + nt * 16 + lhalf;
    float bn = bias[n];
    float vals[8];
#pragma unroll
    for (int v = 0; v < 8; ++v) {
      float val = acc[nt][v] + bn;
      if (relu) val = fmaxf(val, 0.f);
      vals[v] = val * keep[v];       // mask applied after activation (matches ref)
    }
    if (Yf) {
#pragma unroll
      for (int v = 0; v < 8; ++v) Yf[(size_t)(rowBase + v) * N + n] = vals[v];
    }
    if (Yh) {
#pragma unroll
      for (int v = 0; v < 8; ++v) Yh[(size_t)(rowBase + v) * N + n] = (_Float16)vals[v];
    }
  }
}

// ---------------------------------------------------------------------------
// Flash attention, one wave == 16 queries, d = 256, online softmax.
// grid = (NQ/128, B), block = 256 (8 waves).
// ---------------------------------------------------------------------------
__global__ __launch_bounds__(256)
void flash_attn_kernel(const _Float16* __restrict__ Qh, const _Float16* __restrict__ Kh,
                       const _Float16* __restrict__ Vh,
                       const unsigned char* __restrict__ mask,
                       float* __restrict__ O)
{
  __shared__ alignas(16) _Float16 Ks[32][264];     // keys x channels
  __shared__ alignas(16) _Float16 Vs[256][40];     // channels x keys (transposed)
  __shared__ alignas(16) _Float16 Ps[8][16][40];   // per-wave P staging (16x32)

  const int tid   = threadIdx.x;
  const int wv    = tid >> 5;
  const int lane  = tid & 31;
  const int lhalf = lane & 15;
  const int lhi   = lane >> 4;
  const int b     = blockIdx.y;
  const int qb    = blockIdx.x * 128 + wv * 16;

  // resident Q fragments: 16 queries x 256 channels (8 chunks of K=32)
  Frag qf[8];
  {
    const size_t baseQ = ((size_t)b * NQS + qb + lhalf) * CCH;
#pragma unroll
    for (int kk = 0; kk < 8; ++kk) {
      const uint4* p = reinterpret_cast<const uint4*>(Qh + baseQ + kk * 32 + lhi * 8);
      qf[kk].u[0] = p[0]; qf[kk].u[1] = p[2];
    }
  }

  v8f z = {};
  v8f accv[16];
#pragma unroll
  for (int t = 0; t < 16; ++t) accv[t] = z;
  float rmax[8], rsum[8];
#pragma unroll
  for (int v = 0; v < 8; ++v) { rmax[v] = -1e30f; rsum[v] = 0.f; }

  for (int kb = 0; kb < NQS; kb += 32) {
    // cooperative stage: K (key-major) and V (transposed to channel-major)
#pragma unroll
    for (int it = 0; it < 4; ++it) {
      int j = tid + it * 256;              // uint4 index, 1024 total
      int key = j >> 5, c8 = j & 31;
      size_t goff = ((size_t)b * NQS + kb + key) * CCH + c8 * 8;
      H8 hk; hk.u = *reinterpret_cast<const uint4*>(Kh + goff);
      *reinterpret_cast<uint4*>(&Ks[key][c8 * 8]) = hk.u;
      H8 hv; hv.u = *reinterpret_cast<const uint4*>(Vh + goff);
#pragma unroll
      for (int i = 0; i < 8; ++i) Vs[c8 * 8 + i][key] = hv.h[i];
    }
    if (kb + 32 < NQS)
      __builtin_prefetch(Kh + ((size_t)b * NQS + kb + 32) * CCH, 0, 1);
    __syncthreads();

    // S = Q * K^T for two 16-key subtiles
    float svals[2][8], pvals[2][8], chmax[8];
#pragma unroll
    for (int v = 0; v < 8; ++v) chmax[v] = -1e30f;
#pragma unroll
    for (int st = 0; st < 2; ++st) {
      v8f s = z;
#pragma unroll
      for (int kk = 0; kk < 8; ++kk) {
        Frag bk;
        const uint4* p =
            reinterpret_cast<const uint4*>(&Ks[st * 16 + lhalf][kk * 32 + lhi * 8]);
        bk.u[0] = p[0]; bk.u[1] = p[2];
        s = wmma_f16(qf[kk], bk, s);
      }
      int key = kb + st * 16 + lhalf;
      bool kvalid = mask[(size_t)b * NQS + key] != 0;
#pragma unroll
      for (int v = 0; v < 8; ++v) {
        float sv = kvalid ? s[v] * ATT_SCALE : -1e30f;   // masked key column
        svals[st][v] = sv;
        chmax[v] = fmaxf(chmax[v], sv);
      }
    }

    // online softmax (row reductions within 16-lane WMMA groups)
#pragma unroll
    for (int v = 0; v < 8; ++v) {
      float m = chmax[v];
      m = fmaxf(m, __shfl_xor(m, 1, 32));
      m = fmaxf(m, __shfl_xor(m, 2, 32));
      m = fmaxf(m, __shfl_xor(m, 4, 32));
      m = fmaxf(m, __shfl_xor(m, 8, 32));
      float nm   = fmaxf(rmax[v], m);
      float corr = __expf(rmax[v] - nm);
      rmax[v] = nm;
      float ps = 0.f;
#pragma unroll
      for (int st = 0; st < 2; ++st) {
        float p = __expf(svals[st][v] - nm);
        pvals[st][v] = p;
        ps += p;
      }
      ps += __shfl_xor(ps, 1, 32);
      ps += __shfl_xor(ps, 2, 32);
      ps += __shfl_xor(ps, 4, 32);
      ps += __shfl_xor(ps, 8, 32);
      rsum[v] = rsum[v] * corr + ps;
#pragma unroll
      for (int t = 0; t < 16; ++t) accv[t][v] *= corr;
    }

    // D-layout -> A-layout via wave-private LDS round trip
#pragma unroll
    for (int st = 0; st < 2; ++st)
#pragma unroll
      for (int v = 0; v < 8; ++v)
        Ps[wv][v + lhi * 8][st * 16 + lhalf] = (_Float16)pvals[st][v];
    asm volatile("s_wait_dscnt 0" ::: "memory");

    Frag pa;
    {
      const uint4* p = reinterpret_cast<const uint4*>(&Ps[wv][lhalf][lhi * 8]);
      pa.u[0] = p[0]; pa.u[1] = p[2];
    }
    // O += P * V  (K = 32 keys, 16 channel subtiles)
#pragma unroll
    for (int t = 0; t < 16; ++t) {
      Frag vb;
      const uint4* p = reinterpret_cast<const uint4*>(&Vs[t * 16 + lhalf][lhi * 8]);
      vb.u[0] = p[0]; vb.u[1] = p[2];
      accv[t] = wmma_f16(pa, vb, accv[t]);
    }
    __syncthreads();
  }

  // normalize, zero masked query rows (branchless), store f32
  float invq[8];
#pragma unroll
  for (int v = 0; v < 8; ++v) {
    float keep = mask[(size_t)b * NQS + qb + lhi * 8 + v] ? 1.f : 0.f;
    invq[v] = keep / rsum[v];
  }
#pragma unroll
  for (int t = 0; t < 16; ++t) {
    int ch = t * 16 + lhalf;
#pragma unroll
    for (int v = 0; v < 8; ++v) {
      int q = qb + lhi * 8 + v;
      O[((size_t)b * NQS + q) * CCH + ch] = accv[t][v] * invq[v];
    }
  }
}

// ---------------------------------------------------------------------------
// O = LayerNorm(A + Bm) * g + be   (one wave per 256-wide row, float4 I/O)
// ---------------------------------------------------------------------------
__global__ __launch_bounds__(256)
void ln_add_kernel(const float* __restrict__ A, const float* __restrict__ Bm,
                   const float* __restrict__ g, const float* __restrict__ be,
                   float* __restrict__ O)
{
  int row  = blockIdx.x * 8 + (threadIdx.x >> 5);
  int lane = threadIdx.x & 31;
  size_t base = (size_t)row * CCH + lane * 8;

  float v[8];
  {
    float4 a0 = *reinterpret_cast<const float4*>(&A[base]);
    float4 a1 = *reinterpret_cast<const float4*>(&A[base + 4]);
    float4 b0 = *reinterpret_cast<const float4*>(&Bm[base]);
    float4 b1 = *reinterpret_cast<const float4*>(&Bm[base + 4]);
    v[0] = a0.x + b0.x; v[1] = a0.y + b0.y; v[2] = a0.z + b0.z; v[3] = a0.w + b0.w;
    v[4] = a1.x + b1.x; v[5] = a1.y + b1.y; v[6] = a1.z + b1.z; v[7] = a1.w + b1.w;
  }
  float s = 0.f;
#pragma unroll
  for (int i = 0; i < 8; ++i) s += v[i];
#pragma unroll
  for (int m = 1; m < 32; m <<= 1) s += __shfl_xor(s, m, 32);
  float mu = s * (1.f / CCH);
  float q = 0.f;
#pragma unroll
  for (int i = 0; i < 8; ++i) { float d = v[i] - mu; q += d * d; }
#pragma unroll
  for (int m = 1; m < 32; m <<= 1) q += __shfl_xor(q, m, 32);
  float rstd = rsqrtf(q * (1.f / CCH) + EPSF);

  float4 g0 = *reinterpret_cast<const float4*>(&g[lane * 8]);
  float4 g1 = *reinterpret_cast<const float4*>(&g[lane * 8 + 4]);
  float4 e0 = *reinterpret_cast<const float4*>(&be[lane * 8]);
  float4 e1 = *reinterpret_cast<const float4*>(&be[lane * 8 + 4]);
  float4 o0, o1;
  o0.x = (v[0] - mu) * rstd * g0.x + e0.x;
  o0.y = (v[1] - mu) * rstd * g0.y + e0.y;
  o0.z = (v[2] - mu) * rstd * g0.z + e0.z;
  o0.w = (v[3] - mu) * rstd * g0.w + e0.w;
  o1.x = (v[4] - mu) * rstd * g1.x + e1.x;
  o1.y = (v[5] - mu) * rstd * g1.y + e1.y;
  o1.z = (v[6] - mu) * rstd * g1.z + e1.z;
  o1.w = (v[7] - mu) * rstd * g1.w + e1.w;
  *reinterpret_cast<float4*>(&O[base]) = o0;
  *reinterpret_cast<float4*>(&O[base + 4]) = o1;
}

// ---------------------------------------------------------------------------
// a = softmax(H * W55^T + b55) over P=8  (one wave per row, float4 loads)
// ---------------------------------------------------------------------------
__global__ __launch_bounds__(256)
void fc55_softmax_kernel(const float* __restrict__ H, const float* __restrict__ W,
                         const float* __restrict__ bias, float* __restrict__ A)
{
  int row  = blockIdx.x * 8 + (threadIdx.x >> 5);
  int lane = threadIdx.x & 31;
  float hv[8];
  {
    float4 h0 = *reinterpret_cast<const float4*>(&H[(size_t)row * CCH + lane * 8]);
    float4 h1 = *reinterpret_cast<const float4*>(&H[(size_t)row * CCH + lane * 8 + 4]);
    hv[0] = h0.x; hv[1] = h0.y; hv[2] = h0.z; hv[3] = h0.w;
    hv[4] = h1.x; hv[5] = h1.y; hv[6] = h1.z; hv[7] = h1.w;
  }
  float d[PPD];
#pragma unroll
  for (int p = 0; p < PPD; ++p) {
    float4 w0 = *reinterpret_cast<const float4*>(&W[(size_t)p * CCH + lane * 8]);
    float4 w1 = *reinterpret_cast<const float4*>(&W[(size_t)p * CCH + lane * 8 + 4]);
    float s = hv[0] * w0.x + hv[1] * w0.y + hv[2] * w0.z + hv[3] * w0.w
            + hv[4] * w1.x + hv[5] * w1.y + hv[6] * w1.z + hv[7] * w1.w;
#pragma unroll
    for (int m = 1; m < 32; m <<= 1) s += __shfl_xor(s, m, 32);
    d[p] = s + bias[p];
  }
  float mx = d[0];
#pragma unroll
  for (int p = 1; p < PPD; ++p) mx = fmaxf(mx, d[p]);
  float ss = 0.f;
#pragma unroll
  for (int p = 0; p < PPD; ++p) { d[p] = __expf(d[p] - mx); ss += d[p]; }
  float is = 1.f / ss;
  if (lane < PPD) A[(size_t)row * PPD + lane] = d[lane] * is;
}

// ---------------------------------------------------------------------------
__global__ void pool_mean_kernel(const float* __restrict__ H,
                                 const unsigned char* __restrict__ mask,
                                 float* __restrict__ S)
{
  int b = blockIdx.x, c = threadIdx.x;
  __shared__ float invcnt;
  if (threadIdx.x < 32) {
    int cn = 0;
    for (int q = (int)threadIdx.x; q < NQS; q += 32) cn += mask[(size_t)b * NQS + q] ? 1 : 0;
#pragma unroll
    for (int m = 1; m < 32; m <<= 1) cn += __shfl_xor(cn, m, 32);
    if (threadIdx.x == 0) invcnt = 1.f / (float)cn;
  }
  float acc = 0.f;
  for (int q = 0; q < NQS; ++q) acc += H[((size_t)b * NQS + q) * CCH + c];
  __syncthreads();
  S[(size_t)b * CCH + c] = acc * invcnt;
}

__global__ void fc1_kernel(const float* __restrict__ S, const float* __restrict__ W,
                           const float* __restrict__ bias, float* __restrict__ Z)
{
  int gid = blockIdx.x * 256 + threadIdx.x;   // 2048 = 32*64
  int b = gid >> 6, f = gid & 63;
  float s = 0.f;
  for (int c = 0; c < CCH; ++c) s += S[(size_t)b * CCH + c] * W[(size_t)f * CCH + c];
  Z[gid] = s + bias[f];
}

__global__ void bn_relu_kernel(const float* __restrict__ Z0, const float* __restrict__ g,
                               const float* __restrict__ be, float* __restrict__ Z1)
{
  int f = threadIdx.x;                         // 64 features
  float mu = 0.f;
  for (int b = 0; b < BB; ++b) mu += Z0[b * HID + f];
  mu *= (1.f / BB);
  float var = 0.f;
  for (int b = 0; b < BB; ++b) { float d = Z0[b * HID + f] - mu; var += d * d; }
  var *= (1.f / BB);
  float rstd = rsqrtf(var + EPSF);
  for (int b = 0; b < BB; ++b) {
    float zv = (Z0[b * HID + f] - mu) * rstd * g[f] + be[f];
    Z1[b * HID + f] = zv > 0.f ? zv : 0.f;
  }
}

__global__ void fc23_kernel(const float* __restrict__ Z, const float* __restrict__ W2,
                            const float* __restrict__ b2, const float* __restrict__ W3,
                            const float* __restrict__ b3, float* __restrict__ out)
{
  int gid = blockIdx.x * 256 + threadIdx.x;   // 2048 = 2*32*32
  int which = gid >> 10;
  int idx = gid & 1023;
  int b = idx >> 5, zd = idx & 31;
  const float* W  = which ? W3 : W2;
  const float* bi = which ? b3 : b2;
  float s = 0.f;
  for (int j = 0; j < HID; ++j) s += Z[b * HID + j] * W[zd * HID + j];
  out[gid] = s + bi[zd];
}

// ---------------------------------------------------------------------------
extern "C" void kernel_launch(void* const* d_in, const int* in_sizes, int n_in,
                              void* d_out, int out_size, void* d_ws, size_t ws_size,
                              hipStream_t stream)
{
  (void)in_sizes; (void)n_in; (void)out_size; (void)ws_size;
  const float* x = (const float*)d_in[0];
  const unsigned char* mask = (const unsigned char*)d_in[1];
  auto F = [&](int i) { return (const float*)d_in[i]; };

  // workspace carve-out
  char* ws = (char*)d_ws;
  const size_t MC4 = (size_t)MMR * CCH * sizeof(float);
  const size_t MC2 = (size_t)MMR * CCH * sizeof(_Float16);
  float*    hA  = (float*)(ws);
  float*    hB  = (float*)(ws + MC4);
  float*    tmp = (float*)(ws + 2 * MC4);
  _Float16* Qh  = (_Float16*)(ws + 3 * MC4);
  _Float16* Kh  = (_Float16*)(ws + 3 * MC4 + MC2);
  _Float16* Vh  = (_Float16*)(ws + 3 * MC4 + 2 * MC2);
  float*    Sb  = (float*)(ws + 3 * MC4 + 3 * MC2);   // 32x256
  float*    Z0  = Sb + BB * CCH;                      // 32x64
  float*    Z1  = Z0 + BB * HID;                      // 32x64

  float* outA = (float*)d_out + 2 * BB * ZDIM;        // a at offset 2048

  dim3 gGemm(CCH / 64, MMR / 128), bGemm(256);
  dim3 gFlash(NQS / 128, BB), bFlash(256);
  dim3 gRow(MMR / 8), bRow(256);

  auto gemm = [&](const float* X, const float* W, const float* bia,
                  float* Yf, _Float16* Yh, int relu) {
    gemm_wmma_kernel<<<gGemm, bGemm, 0, stream>>>(X, W, bia, mask, Yf, Yh,
                                                  MMR, CCH, CCH, relu);
  };

  float* h  = hA;
  float* ho = hB;
  auto sa_block = [&](int base) {   // base = index of sa{i}_wq
    gemm(h, F(base + 0), F(base + 1), nullptr, Qh, 0);
    gemm(h, F(base + 2), F(base + 3), nullptr, Kh, 0);
    gemm(h, F(base + 4), F(base + 5), nullptr, Vh, 0);
    flash_attn_kernel<<<gFlash, bFlash, 0, stream>>>(Qh, Kh, Vh, mask, tmp);
    ln_add_kernel<<<gRow, bRow, 0, stream>>>(h, tmp, F(base + 6), F(base + 7), ho);
    float* t = h; h = ho; ho = t;
  };
  auto ffn_block = [&](int base) {  // base = index of fc{nm}_w
    gemm(h, F(base + 0), F(base + 1), tmp, nullptr, 1);
    ln_add_kernel<<<gRow, bRow, 0, stream>>>(tmp, h, F(base + 2), F(base + 3), ho);
    float* t = h; h = ho; ho = t;
  };

  // ---- encoder_a: sa3/fc33, sa4/fc44, sa5, fc55 ----
  hipMemcpyAsync(h, x, MC4, hipMemcpyDeviceToDevice, stream);
  sa_block(2 + 2 * 8);        // sa3
  ffn_block(42 + 2 * 4);      // fc33/ln33
  sa_block(2 + 3 * 8);        // sa4
  ffn_block(42 + 3 * 4);      // fc44/ln44
  sa_block(2 + 4 * 8);        // sa5
  fc55_softmax_kernel<<<gRow, bRow, 0, stream>>>(h, F(58), F(59), outA);

  // ---- encoder_z: sa1/fc11, sa2/fc22 ----
  hipMemcpyAsync(h, x, MC4, hipMemcpyDeviceToDevice, stream);
  sa_block(2 + 0 * 8);        // sa1
  ffn_block(42 + 0 * 4);      // fc11/ln11
  sa_block(2 + 1 * 8);        // sa2
  ffn_block(42 + 1 * 4);      // fc22/ln22

  pool_mean_kernel<<<dim3(BB), dim3(CCH), 0, stream>>>(h, mask, Sb);
  fc1_kernel<<<dim3((BB * HID) / 256), dim3(256), 0, stream>>>(Sb, F(60), F(61), Z0);
  bn_relu_kernel<<<dim3(1), dim3(HID), 0, stream>>>(Z0, F(62), F(63), Z1);
  fc23_kernel<<<dim3((2 * BB * ZDIM) / 256), dim3(256), 0, stream>>>(
      Z1, F(64), F(65), F(66), F(67), (float*)d_out);
}
